// LSTM_1073741824385
// MI455X (gfx1250) — matrix operand order repaired and checked by reading, compiled, not run on hardware
//
#include <hip/hip_runtime.h>
#include <hip/hip_bf16.h>

// ---------------------------------------------------------------------------
// 2-layer LSTM (B=32,S=512,IN=256,H=512) + Linear(OUT=256) for gfx1250.
// All matmuls via v_wmma_f32_16x16x32_bf16 (fp32 accumulate, bf16 operands).
// ---------------------------------------------------------------------------

#define KB 32
#define KS 512
#define KIN 256
#define KH 512
#define KOUT 256
#define K4H (4 * KH)

typedef __attribute__((ext_vector_type(16))) __bf16 v16bf;
typedef __attribute__((ext_vector_type(8)))  float  v8f;

union FragU { v16bf v; uint4 q[2]; };

__device__ __forceinline__ unsigned short f2bf(float f) {
    unsigned u = __builtin_bit_cast(unsigned, f);
    unsigned r = u + 0x7FFFu + ((u >> 16) & 1u);   // round-to-nearest-even
    return (unsigned short)(r >> 16);
}

// A-matrix 16x32 bf16 fragment (rows r0..r0+15 of row-major [*, ldk]).
// ISA layout: lanes 0-15 hold K {k0..k0+7, k0+16..k0+23}; lanes 16-31 same +8.
__device__ __forceinline__ v16bf load_a_frag(const unsigned short* __restrict__ base,
                                             int ldk, int r0, int k0, int lane) {
    int r = r0 + (lane & 15);
    int k = k0 + ((lane >> 4) << 3);
    const unsigned short* p = base + (size_t)r * ldk + k;
    FragU f;
    f.q[0] = *(const uint4*)(p);        // K = k .. k+7
    f.q[1] = *(const uint4*)(p + 16);   // K = k+16 .. k+23
    return f.v;
}

// B-matrix 32x16 bf16 fragment taken from row-major weights W[N,K] (B = W^T).
// ISA layout: lanes 0-15 hold K {k0..k0+7, k0+16..k0+23}; lanes 16-31 same +8.
__device__ __forceinline__ v16bf load_b_frag(const unsigned short* __restrict__ Wt,
                                             int ldk, int n0, int k0, int lane) {
    int n = n0 + (lane & 15);
    int k = k0 + ((lane >> 4) << 3);
    const unsigned short* p = Wt + (size_t)n * ldk + k;
    FragU f;
    f.q[0] = *(const uint4*)(p);        // K = k .. k+7
    f.q[1] = *(const uint4*)(p + 16);   // K = k+16 .. k+23
    return f.v;
}

__device__ __forceinline__ v8f wmma_bf16(v16bf a, v16bf b, v8f c) {
    return __builtin_amdgcn_wmma_f32_16x16x32_bf16(false, a, false, b,
                                                   (short)0, c, false, false);
}

// --------------------------- conversion kernels ----------------------------

__global__ void conv_f32_to_bf16(const float* __restrict__ src,
                                 unsigned short* __restrict__ dst, int n) {
    int i = blockIdx.x * blockDim.x + threadIdx.x;
    if (i < n) dst[i] = f2bf(src[i]);
}

// x: [B,S,IN] fp32  ->  x_bf: [S,B,IN] bf16 (time-major for the big GEMM)
__global__ void conv_x_time_major(const float* __restrict__ x,
                                  unsigned short* __restrict__ dst) {
    int i = blockIdx.x * blockDim.x + threadIdx.x;
    if (i >= KB * KS * KIN) return;
    int k = i % KIN;
    int s = (i / KIN) % KS;
    int b = i / (KIN * KS);
    dst[((size_t)s * KB + b) * KIN + k] = f2bf(x[i]);
}

__global__ void zero_state(unsigned short* __restrict__ hbf,
                           float* __restrict__ c, int n) {
    int i = blockIdx.x * blockDim.x + threadIdx.x;
    if (i < n) { hbf[i] = 0; c[i] = 0.0f; }
}

// ------------------------------ GEMM kernel --------------------------------
// C[M,N] = A[M,K](bf16) * W[N,K]^T(bf16) + bias1[n] + bias2[n]   (fp32 out)
// Block: 256 threads = 8 waves in a 2x4 grid; block tile 64(M) x 64(N);
// each wave produces a 32x16 tile with two 16x16x32 bf16 WMMAs per k-step.
// out_transpose: row m=(s*B+b) is written to C[(b*S+s)*N + n] (linear head).
__global__ __launch_bounds__(256) void gemm_bf16(
    const unsigned short* __restrict__ A, const unsigned short* __restrict__ Wt,
    const float* __restrict__ bias1, const float* __restrict__ bias2,
    float* __restrict__ C, int M, int N, int K,
    int out_transpose, int Bdim, int Sdim) {
    const int lane = threadIdx.x & 31;
    const int wave = threadIdx.x >> 5;
    const int m_base = blockIdx.x * 64 + (wave >> 2) * 32;
    const int n_base = blockIdx.y * 64 + (wave & 3) * 16;

    v8f acc0 = {}; v8f acc1 = {};
    for (int k0 = 0; k0 < K; k0 += 32) {
        v16bf b  = load_b_frag(Wt, K, n_base, k0, lane);
        v16bf a0 = load_a_frag(A, K, m_base,      k0, lane);
        v16bf a1 = load_a_frag(A, K, m_base + 16, k0, lane);
        acc0 = wmma_bf16(a0, b, acc0);
        acc1 = wmma_bf16(a1, b, acc1);
    }

    const int n = n_base + (lane & 15);
    float bias = (bias1 ? bias1[n] : 0.0f) + (bias2 ? bias2[n] : 0.0f);
    const int mo = (lane >> 4) << 3;
#pragma unroll
    for (int mt = 0; mt < 2; ++mt) {
        v8f acc = mt ? acc1 : acc0;
#pragma unroll
        for (int r = 0; r < 8; ++r) {
            int m = m_base + mt * 16 + r + mo;
            float v = acc[r] + bias;
            size_t oi;
            if (!out_transpose) {
                oi = (size_t)m * N + n;
            } else {
                int s = m / Bdim, bb = m % Bdim;
                oi = ((size_t)bb * Sdim + s) * N + n;
            }
            C[oi] = v;
        }
    }
}

// --------------------------- LSTM step kernel ------------------------------
// One timestep. Block i owns h-columns [32i, 32i+32) -> gate columns
// {q*H + 32i .. +32} for q = i,f,g,o (block-local 128 cols). 16 blocks/step.
// GEMM: gates[32,128] = h_prev[32,512](bf16) * Whh_cols^T, + xp, then the
// elementwise cell update via LDS. h written as bf16 into h_seq[t] (which is
// disjoint from h_seq[t-1] -> no cross-block race).
__global__ __launch_bounds__(256) void lstm_step(
    const float* __restrict__ xp,              // [S*B, 4H], row = t*B + b
    const unsigned short* __restrict__ Whh,    // [4H, H] bf16 row-major
    const unsigned short* __restrict__ h_prev, // [B, H] bf16
    float* __restrict__ c_state,               // [B, H] fp32 (in place)
    unsigned short* __restrict__ h_out,        // [B, H] bf16 = h_seq[t]
    float* __restrict__ hn_out, float* __restrict__ cn_out,
    int t, int is_last) {
    __shared__ float gl[32 * 128];

    const int lane = threadIdx.x & 31;
    const int wave = threadIdx.x >> 5;         // n-tile 0..7
    const int h0   = blockIdx.x * 32;
    const int q    = wave >> 1;                // gate 0..3 (i,f,g,o)
    const int csub = (wave & 1) * 16;
    const int wrow0 = q * KH + h0 + csub;      // W_hh row base for this tile

    v8f acc0 = {}; v8f acc1 = {};
    for (int k0 = 0; k0 < KH; k0 += 32) {
        v16bf b  = load_b_frag(Whh, KH, wrow0, k0, lane);
        v16bf a0 = load_a_frag(h_prev, KH, 0,  k0, lane);
        v16bf a1 = load_a_frag(h_prev, KH, 16, k0, lane);
        acc0 = wmma_bf16(a0, b, acc0);
        acc1 = wmma_bf16(a1, b, acc1);
    }

    // add precomputed input projection, stage gates in LDS
    const int nl   = wave * 16 + (lane & 15);            // block-local col
    const int gcol = q * KH + h0 + csub + (lane & 15);   // global gate col
    const int mo   = (lane >> 4) << 3;
    const float* xprow = xp + ((size_t)t * KB) * K4H;
#pragma unroll
    for (int mt = 0; mt < 2; ++mt) {
        v8f acc = mt ? acc1 : acc0;
#pragma unroll
        for (int r = 0; r < 8; ++r) {
            int b = mt * 16 + r + mo;                    // batch row 0..31
            gl[b * 128 + nl] = acc[r] + xprow[(size_t)b * K4H + gcol];
        }
    }
    __syncthreads();

    // cell update: 32 batches x 32 h-cols = 1024 elems over 256 threads
#pragma unroll
    for (int it = 0; it < 4; ++it) {
        int e = threadIdx.x + it * 256;
        int b = e >> 5, c = e & 31;
        float vi = gl[b * 128 +       c];
        float vf = gl[b * 128 +  32 + c];
        float vg = gl[b * 128 +  64 + c];
        float vo = gl[b * 128 +  96 + c];
        vi = 1.0f / (1.0f + __expf(-vi));
        vf = 1.0f / (1.0f + __expf(-vf));
        vg = tanhf(vg);
        vo = 1.0f / (1.0f + __expf(-vo));
        int idx = b * KH + h0 + c;
        float cn = vf * c_state[idx] + vi * vg;
        float hn = vo * tanhf(cn);
        c_state[idx] = cn;
        h_out[idx] = f2bf(hn);
        if (is_last) { hn_out[idx] = hn; cn_out[idx] = cn; }
    }
}

// ------------------------------- launcher ----------------------------------

extern "C" void kernel_launch(void* const* d_in, const int* in_sizes, int n_in,
                              void* d_out, int out_size, void* d_ws, size_t ws_size,
                              hipStream_t stream) {
    (void)in_sizes; (void)n_in; (void)out_size; (void)ws_size;

    const float* x     = (const float*)d_in[0];
    const float* W_ih0 = (const float*)d_in[1];
    const float* W_hh0 = (const float*)d_in[2];
    const float* b_ih0 = (const float*)d_in[3];
    const float* b_hh0 = (const float*)d_in[4];
    const float* W_ih1 = (const float*)d_in[5];
    const float* W_hh1 = (const float*)d_in[6];
    const float* b_ih1 = (const float*)d_in[7];
    const float* b_hh1 = (const float*)d_in[8];
    const float* W_lin = (const float*)d_in[9];
    const float* b_lin = (const float*)d_in[10];
    float* out = (float*)d_out;

    // workspace carve-out (256B aligned)
    char* wp = (char*)d_ws;
    auto carve = [&](size_t bytes) -> void* {
        void* r = (void*)wp;
        wp += (bytes + 255) & ~(size_t)255;
        return r;
    };
    unsigned short* x_bf  = (unsigned short*)carve((size_t)KB * KS * KIN * 2);
    unsigned short* Wih0b = (unsigned short*)carve((size_t)K4H * KIN * 2);
    unsigned short* Whh0b = (unsigned short*)carve((size_t)K4H * KH * 2);
    unsigned short* Wih1b = (unsigned short*)carve((size_t)K4H * KH * 2);
    unsigned short* Whh1b = (unsigned short*)carve((size_t)K4H * KH * 2);
    unsigned short* Wlinb = (unsigned short*)carve((size_t)KOUT * KH * 2);
    unsigned short* h1seq = (unsigned short*)carve((size_t)KS * KB * KH * 2);
    unsigned short* h2seq = (unsigned short*)carve((size_t)KS * KB * KH * 2);
    unsigned short* h0z   = (unsigned short*)carve((size_t)KB * KH * 2);
    float* c0 = (float*)carve((size_t)KB * KH * 4);
    float* c1 = (float*)carve((size_t)KB * KH * 4);
    float* xp = (float*)carve((size_t)KS * KB * K4H * 4);   // reused by layer 1

    const int M  = KS * KB;        // 16384
    float* hn_base = out + (size_t)KB * KS * KOUT;          // h_n [2,B,H]
    float* cn_base = hn_base + 2 * KB * KH;                 // c_n [2,B,H]

    // 1) precision conversion
    conv_x_time_major<<<(KB * KS * KIN + 255) / 256, 256, 0, stream>>>(x, x_bf);
    conv_f32_to_bf16<<<(K4H * KIN + 255) / 256, 256, 0, stream>>>(W_ih0, Wih0b, K4H * KIN);
    conv_f32_to_bf16<<<(K4H * KH + 255) / 256, 256, 0, stream>>>(W_hh0, Whh0b, K4H * KH);
    conv_f32_to_bf16<<<(K4H * KH + 255) / 256, 256, 0, stream>>>(W_ih1, Wih1b, K4H * KH);
    conv_f32_to_bf16<<<(K4H * KH + 255) / 256, 256, 0, stream>>>(W_hh1, Whh1b, K4H * KH);
    conv_f32_to_bf16<<<(KOUT * KH + 255) / 256, 256, 0, stream>>>(W_lin, Wlinb, KOUT * KH);
    zero_state<<<(KB * KH + 255) / 256, 256, 0, stream>>>(h0z, c0, KB * KH);
    zero_state<<<(KB * KH + 255) / 256, 256, 0, stream>>>(h0z, c1, KB * KH);

    // 2) layer-0 input projection: xp = x_bf * W_ih0^T + b_ih0 + b_hh0
    gemm_bf16<<<dim3(M / 64, K4H / 64), 256, 0, stream>>>(
        x_bf, Wih0b, b_ih0, b_hh0, xp, M, K4H, KIN, 0, KB, KS);

    // 3) layer-0 recurrence
    for (int t = 0; t < KS; ++t) {
        const unsigned short* hp = (t == 0) ? h0z : h1seq + (size_t)(t - 1) * KB * KH;
        lstm_step<<<KH / 32, 256, 0, stream>>>(
            xp, Whh0b, hp, c0, h1seq + (size_t)t * KB * KH,
            hn_base, cn_base, t, t == KS - 1);
    }

    // 4) layer-1 input projection: xp = h1seq * W_ih1^T + b_ih1 + b_hh1
    gemm_bf16<<<dim3(M / 64, K4H / 64), 256, 0, stream>>>(
        h1seq, Wih1b, b_ih1, b_hh1, xp, M, K4H, KH, 0, KB, KS);

    // 5) layer-1 recurrence
    for (int t = 0; t < KS; ++t) {
        const unsigned short* hp = (t == 0) ? h0z : h2seq + (size_t)(t - 1) * KB * KH;
        lstm_step<<<KH / 32, 256, 0, stream>>>(
            xp, Whh1b, hp, c1, h2seq + (size_t)t * KB * KH,
            hn_base + KB * KH, cn_base + KB * KH, t, t == KS - 1);
    }

    // 6) linear head: out[b,s,:] = h2[s,b,:] * W_lin^T + b_lin
    gemm_bf16<<<dim3(M / 64, KOUT / 64), 256, 0, stream>>>(
        h2seq, Wlinb, b_lin, nullptr, out, M, KOUT, KH, 1, KB, KS);
}
